// StyleGAN2Generator_13915694039324
// MI455X (gfx1250) — matrix operand order repaired and checked by reading, not compiled, not run on hardware
//
#include <hip/hip_runtime.h>
#include <hip/hip_fp16.h>

typedef __attribute__((ext_vector_type(16))) _Float16 v16h;
typedef __attribute__((ext_vector_type(8)))  _Float16 v8h;
typedef __attribute__((ext_vector_type(8)))  float    v8f;

#define BATCH 4
#define STYLE_DIM 512
#define SLOPE 0.2f
#define GAIN 1.41421356237309515f

// ---------------------------------------------------------------------------
// Mapping network: 8x EqualLinear + leaky_relu, one workgroup, LDS-resident.
// ---------------------------------------------------------------------------
struct MapArgs {
    const float* W[8];
    const float* Bb[8];
};

__global__ void mapping_kernel(const float* __restrict__ z,
                               const float* __restrict__ cfeat,
                               MapArgs ma,
                               float* __restrict__ wlat) {
    __shared__ float act[BATCH * 1024];
    int tid = threadIdx.x;  // 512 threads
    for (int b = 0; b < BATCH; ++b) {
        act[b * 1024 + tid]       = z[b * 512 + tid];
        act[b * 1024 + 512 + tid] = cfeat[b * 512 + tid];
    }
    __syncthreads();
    int din = 1024;
    for (int l = 0; l < 8; ++l) {
        const float* Wl = ma.W[l];
        float scale = rsqrtf((float)din);
        float bias  = ma.Bb[l][tid];
        float s0 = 0.f, s1 = 0.f, s2 = 0.f, s3 = 0.f;
        for (int k = 0; k < din; ++k) {
            float wv = Wl[tid * din + k];
            s0 += act[0 * 1024 + k] * wv;
            s1 += act[1 * 1024 + k] * wv;
            s2 += act[2 * 1024 + k] * wv;
            s3 += act[3 * 1024 + k] * wv;
        }
        s0 = s0 * scale + bias; s1 = s1 * scale + bias;
        s2 = s2 * scale + bias; s3 = s3 * scale + bias;
        s0 = (s0 < 0.f) ? s0 * SLOPE : s0;
        s1 = (s1 < 0.f) ? s1 * SLOPE : s1;
        s2 = (s2 < 0.f) ? s2 * SLOPE : s2;
        s3 = (s3 < 0.f) ? s3 * SLOPE : s3;
        __syncthreads();
        act[0 * 1024 + tid] = s0;
        act[1 * 1024 + tid] = s1;
        act[2 * 1024 + tid] = s2;
        act[3 * 1024 + tid] = s3;
        __syncthreads();
        din = 512;
    }
    for (int b = 0; b < BATCH; ++b)
        wlat[b * 512 + tid] = act[b * 1024 + tid];
}

// ---------------------------------------------------------------------------
// Style EqualLinear: s[b,i] = (wlat[b,:] . mw[i,:]) / sqrt(512) + mb[i]
// ---------------------------------------------------------------------------
__global__ void style_kernel(const float* __restrict__ wlat,
                             const float* __restrict__ mw,
                             const float* __restrict__ mb,
                             float* __restrict__ s, int chans) {
    int i = blockIdx.x * blockDim.x + threadIdx.x;
    if (i >= chans) return;
    float scale = rsqrtf((float)STYLE_DIM);
    float bias = mb[i];
    for (int b = 0; b < BATCH; ++b) {
        float acc = 0.f;
        const float* wr = wlat + b * STYLE_DIM;
        const float* mr = mw + (size_t)i * STYLE_DIM;
        for (int k = 0; k < STYLE_DIM; ++k) acc += wr[k] * mr[k];
        s[b * chans + i] = acc * scale + bias;
    }
}

// ---------------------------------------------------------------------------
// Demodulation factor: d[b,o] = rsqrt( sum_i s^2 * sum_t (w*cs)^2 + eps )
// ---------------------------------------------------------------------------
__global__ void demod_kernel(const float* __restrict__ s,
                             const float* __restrict__ w,
                             float* __restrict__ dmod, int I, int O) {
    int idx = blockIdx.x * blockDim.x + threadIdx.x;
    if (idx >= BATCH * O) return;
    int b = idx / O, o = idx % O;
    float cs2 = 1.0f / ((float)I * 9.0f);  // (1/sqrt(I*9))^2
    const float* wo = w + (size_t)o * I * 9;
    float acc = 0.f;
    for (int i = 0; i < I; ++i) {
        float sv = s[b * I + i];
        float w2 = 0.f;
        for (int t = 0; t < 9; ++t) { float wv = wo[i * 9 + t]; w2 += wv * wv; }
        acc += sv * sv * w2;
    }
    dmod[idx] = rsqrtf(acc * cs2 + 1e-8f);
}

// ---------------------------------------------------------------------------
// Pack conv weights to per-tap f16 GEMM layout wh[t][o][i] = w[o][i][t]*cs.
// ---------------------------------------------------------------------------
__global__ void prep_w_kernel(const float* __restrict__ w,
                              _Float16* __restrict__ wh, int I, int O) {
    int idx = blockIdx.x * blockDim.x + threadIdx.x;
    if (idx >= O * I * 9) return;
    int o   = idx / (I * 9);
    int rem = idx - o * I * 9;
    int i   = rem / 9;
    int t   = rem - i * 9;
    float v = w[idx] * rsqrtf((float)I * 9.0f);
    wh[((size_t)t * O + o) * I + i] = (_Float16)v;
}

// ---------------------------------------------------------------------------
// Modulate input -> f16 NHWC with 1-pixel zero halo: xs[b][(H+2)(W+2)][I].
// Optionally 2x-nearest upsample of NCHW source, or broadcast constant.
// ---------------------------------------------------------------------------
__global__ void modulate_pad_kernel(const float* __restrict__ x,
                                    const float* __restrict__ cst,
                                    const float* __restrict__ s,
                                    _Float16* __restrict__ xs,
                                    int I, int H, int W, int ups, int from_const) {
    int PH = H + 2, PW = W + 2;
    long idx = (long)blockIdx.x * blockDim.x + threadIdx.x;
    long total = (long)BATCH * PH * PW * I;
    if (idx >= total) return;
    int  i  = (int)(idx % I);
    long pp = (idx / I) % ((long)PH * PW);
    int  b  = (int)(idx / ((long)I * PH * PW));
    int  y  = (int)(pp / PW) - 1;
    int  xp = (int)(pp % PW) - 1;
    float v = 0.f;
    if (y >= 0 && y < H && xp >= 0 && xp < W) {
        if (from_const) {
            v = cst[(size_t)i * H * W + (size_t)y * W + xp];
        } else if (ups) {
            int Hh = H >> 1, Wh = W >> 1;
            v = x[((size_t)(b * I + i) * Hh + (y >> 1)) * Wh + (xp >> 1)];
        } else {
            v = x[(size_t)(b * I + i) * H * W + (size_t)y * W + xp];
        }
        v *= s[b * I + i];
    }
    xs[idx] = (_Float16)v;
}

// ---------------------------------------------------------------------------
// Fragment loaders (both operands are two contiguous 16B loads per lane).
// A (16x32, lane row = lane&15): halves {k0..k0+7, k0+16..k0+23} (+8 for hi).
// B (32x16, lane col = lane&15): same halves as A, {k0..k0+7, k0+16..k0+23} (+8 for hi).
// ---------------------------------------------------------------------------
__device__ __forceinline__ v16h load_afrag(const _Float16* __restrict__ a8) {
    v8h al = *reinterpret_cast<const v8h*>(a8);
    v8h ah = *reinterpret_cast<const v8h*>(a8 + 16);
    v16h a;
#pragma unroll
    for (int j = 0; j < 8; ++j) { a[j] = al[j]; a[j + 8] = ah[j]; }
    return a;
}

__device__ __forceinline__ v16h load_bfrag(const _Float16* __restrict__ b16) {
    v8h bl = *reinterpret_cast<const v8h*>(b16);
    v8h bh = *reinterpret_cast<const v8h*>(b16 + 16);
    v16h bf;
#pragma unroll
    for (int j = 0; j < 8; ++j) { bf[j] = bl[j]; bf[j + 8] = bh[j]; }
    return bf;
}

__device__ __forceinline__ v8f wmma_acc(v16h a, v16h bf, v8f acc) {
    return __builtin_amdgcn_wmma_f32_16x16x32_f16(
        false, a, false, bf, (short)0, acc, false, false);
}

__device__ __forceinline__ float conv_epilogue(float acc, float d, float nz,
                                               float bias) {
    float v = acc * d + nz + bias;
    return ((v < 0.f) ? v * SLOPE : v) * GAIN;
}

// ---------------------------------------------------------------------------
// Implicit-GEMM 3x3 modulated conv, 16(O) x 16(pix) per wave (small res).
// ---------------------------------------------------------------------------
__global__ void conv3x3_wmma16_kernel(const _Float16* __restrict__ xs,  // [B,(H+2)(W+2),I]
                                      const _Float16* __restrict__ wh,  // [9,O,I]
                                      const float* __restrict__ dmod,   // [B,O]
                                      const float* __restrict__ noise,  // [B,H,W]
                                      const float* __restrict__ nw,     // scalar
                                      const float* __restrict__ bias,   // [O]
                                      float* __restrict__ out,          // [B,O,H,W]
                                      int I, int O, int H, int W) {
    int lane   = threadIdx.x & 31;
    int laneLo = lane & 15;
    int laneHi = lane >> 4;
    int b      = blockIdx.z;
    int oTile  = blockIdx.y;
    int p      = blockIdx.x * 16 + laneLo;
    int PW     = W + 2;
    int npix   = H * W;
    int py     = p / W, px = p % W;

    const _Float16* xsb   = xs + (size_t)b * (H + 2) * PW * I;
    const _Float16* wbase = wh + ((size_t)oTile * 16 + laneLo) * I + laneHi * 8;
    size_t tapStride = (size_t)O * I;

    v8f acc0 = {}, acc1 = {};
#pragma unroll
    for (int t = 0; t < 9; ++t) {
        int dy = t / 3, dx = t - (t / 3) * 3;
        const _Float16* bbase = xsb + ((size_t)(py + dy) * PW + (px + dx)) * I + laneHi * 8;
        const _Float16* abase = wbase + (size_t)t * tapStride;
        for (int k0 = 0; k0 < I; k0 += 64) {
            acc0 = wmma_acc(load_afrag(abase + k0),      load_bfrag(bbase + k0),      acc0);
            acc1 = wmma_acc(load_afrag(abase + k0 + 32), load_bfrag(bbase + k0 + 32), acc1);
        }
    }
#pragma unroll
    for (int e = 0; e < 8; ++e) acc0[e] += acc1[e];

    float nz0 = nw[0] * noise[(size_t)b * npix + p];
#pragma unroll
    for (int e = 0; e < 8; ++e) {
        int o = oTile * 16 + laneHi * 8 + e;
        out[((size_t)(b * O + o)) * npix + p] =
            conv_epilogue(acc0[e], dmod[b * O + o], nz0, bias[o]);
    }
}

// ---------------------------------------------------------------------------
// Implicit-GEMM 3x3 modulated conv, 32(O) x 32(pix) register tile per wave:
// 4 WMMAs per 8 b128 loads (2x the FLOP/byte), 4 independent acc chains.
// Requires npix % 32 == 0 and O % 32 == 0 (true for r >= 8).
// ---------------------------------------------------------------------------
__global__ void conv3x3_wmma32_kernel(const _Float16* __restrict__ xs,  // [B,(H+2)(W+2),I]
                                      const _Float16* __restrict__ wh,  // [9,O,I]
                                      const float* __restrict__ dmod,   // [B,O]
                                      const float* __restrict__ noise,  // [B,H,W]
                                      const float* __restrict__ nw,     // scalar
                                      const float* __restrict__ bias,   // [O]
                                      float* __restrict__ out,          // [B,O,H,W]
                                      int I, int O, int H, int W) {
    int lane   = threadIdx.x & 31;
    int laneLo = lane & 15;
    int laneHi = lane >> 4;
    int b      = blockIdx.z;
    int o0     = blockIdx.y * 32;
    int p0     = blockIdx.x * 32;
    int PW     = W + 2;
    int npix   = H * W;

    int pA = p0 + laneLo;                // B0 column pixel
    int pB = p0 + 16 + laneLo;           // B1 column pixel
    int pyA = pA / W, pxA = pA % W;
    int pyB = pB / W, pxB = pB % W;

    const _Float16* xsb    = xs + (size_t)b * (H + 2) * PW * I;
    const _Float16* wbase0 = wh + ((size_t)o0 + laneLo) * I + laneHi * 8;       // A0 rows
    const _Float16* wbase1 = wbase0 + (size_t)16 * I;                           // A1 rows
    size_t tapStride = (size_t)O * I;

    v8f acc00 = {}, acc01 = {}, acc10 = {}, acc11 = {};
#pragma unroll
    for (int t = 0; t < 9; ++t) {
        int dy = t / 3, dx = t - (t / 3) * 3;
        const _Float16* bbA = xsb + ((size_t)(pyA + dy) * PW + (pxA + dx)) * I + laneHi * 8;
        const _Float16* bbB = xsb + ((size_t)(pyB + dy) * PW + (pxB + dx)) * I + laneHi * 8;
        const _Float16* ab0 = wbase0 + (size_t)t * tapStride;
        const _Float16* ab1 = wbase1 + (size_t)t * tapStride;
        for (int k0 = 0; k0 < I; k0 += 32) {
            v16h a0 = load_afrag(ab0 + k0);
            v16h a1 = load_afrag(ab1 + k0);
            v16h b0 = load_bfrag(bbA + k0);
            v16h b1 = load_bfrag(bbB + k0);
            acc00 = wmma_acc(a0, b0, acc00);
            acc01 = wmma_acc(a0, b1, acc01);
            acc10 = wmma_acc(a1, b0, acc10);
            acc11 = wmma_acc(a1, b1, acc11);
        }
    }

    float nw0 = nw[0];
    float nzA = nw0 * noise[(size_t)b * npix + pA];
    float nzB = nw0 * noise[(size_t)b * npix + pB];
#pragma unroll
    for (int e = 0; e < 8; ++e) {
        int oA = o0 + laneHi * 8 + e;        // rows of A0 tile
        int oB = oA + 16;                    // rows of A1 tile
        float dA = dmod[b * O + oA], biA = bias[oA];
        float dB = dmod[b * O + oB], biB = bias[oB];
        out[((size_t)(b * O + oA)) * npix + pA] = conv_epilogue(acc00[e], dA, nzA, biA);
        out[((size_t)(b * O + oA)) * npix + pB] = conv_epilogue(acc01[e], dA, nzB, biA);
        out[((size_t)(b * O + oB)) * npix + pA] = conv_epilogue(acc10[e], dB, nzA, biB);
        out[((size_t)(b * O + oB)) * npix + pB] = conv_epilogue(acc11[e], dB, nzB, biB);
    }
}

// ---------------------------------------------------------------------------
// to_rgb: 1x1 modulated conv (no demod) + bias + upsampled skip.
// ---------------------------------------------------------------------------
__global__ void torgb_kernel(const float* __restrict__ x,    // [B,O,H,W]
                             const float* __restrict__ s,    // [B,O]
                             const float* __restrict__ wr,   // [3,O]
                             const float* __restrict__ br,   // [3]
                             const float* __restrict__ skip, // [B,3,H/2,W/2] or null
                             float* __restrict__ rgb,        // [B,3,H,W]
                             int O, int H, int W, int has_skip) {
    long idx = (long)blockIdx.x * blockDim.x + threadIdx.x;
    long npix = (long)H * W;
    if (idx >= (long)BATCH * npix) return;
    long p = idx % npix;
    int  b = (int)(idx / npix);
    float scale = rsqrtf((float)O);
    float a0 = 0.f, a1 = 0.f, a2 = 0.f;
    for (int o = 0; o < O; ++o) {
        float xv = x[(size_t)(b * O + o) * npix + p] * s[b * O + o];
        a0 += wr[0 * O + o] * xv;
        a1 += wr[1 * O + o] * xv;
        a2 += wr[2 * O + o] * xv;
    }
    a0 = a0 * scale + br[0];
    a1 = a1 * scale + br[1];
    a2 = a2 * scale + br[2];
    if (has_skip) {
        int y = (int)(p / W), xx = (int)(p % W);
        int Hh = H >> 1, Wh = W >> 1;
        size_t sp = ((size_t)y >> 1) * Wh + (xx >> 1);
        a0 += skip[((size_t)(b * 3 + 0) * Hh) * Wh + sp];
        a1 += skip[((size_t)(b * 3 + 1) * Hh) * Wh + sp];
        a2 += skip[((size_t)(b * 3 + 2) * Hh) * Wh + sp];
    }
    rgb[(size_t)(b * 3 + 0) * npix + p] = a0;
    rgb[(size_t)(b * 3 + 1) * npix + p] = a1;
    rgb[(size_t)(b * 3 + 2) * npix + p] = a2;
}

__global__ void tanh_kernel(const float* __restrict__ in, float* __restrict__ out, long n) {
    long idx = (long)blockIdx.x * blockDim.x + threadIdx.x;
    if (idx < n) out[idx] = tanhf(in[idx]);
}

// ---------------------------------------------------------------------------
// Host orchestration
// ---------------------------------------------------------------------------
static inline void launch_conv(const _Float16* xs, const _Float16* wh,
                               const float* dmod, const float* nz, const float* nwp,
                               const float* bs, float* out,
                               int I, int O, int r, hipStream_t stream) {
    int npix = r * r;
    if ((npix % 32 == 0) && (O % 32 == 0)) {
        dim3 g(npix / 32, O / 32, BATCH);
        conv3x3_wmma32_kernel<<<g, 32, 0, stream>>>(xs, wh, dmod, nz, nwp, bs, out,
                                                    I, O, r, r);
    } else {
        dim3 g(npix / 16, O / 16, BATCH);
        conv3x3_wmma16_kernel<<<g, 32, 0, stream>>>(xs, wh, dmod, nz, nwp, bs, out,
                                                    I, O, r, r);
    }
}

extern "C" void kernel_launch(void* const* d_in, const int* in_sizes, int n_in,
                              void* d_out, int out_size, void* d_ws, size_t ws_size,
                              hipStream_t stream) {
    (void)in_sizes; (void)n_in; (void)out_size; (void)ws_size;
    static const int RESV[7] = {4, 8, 16, 32, 64, 128, 256};
    static const int CHV[7]  = {512, 512, 512, 512, 256, 128, 64};

    const float* z  = (const float*)d_in[0];
    const float* cf = (const float*)d_in[1];
    MapArgs ma;
    for (int l = 0; l < 8; ++l) {
        ma.W[l]  = (const float*)d_in[2 + l];
        ma.Bb[l] = (const float*)d_in[10 + l];
    }
    const float* constant = (const float*)d_in[18];
    // conv params base 19 + idx*10: [w1,mw1,mb1,nw1,b1, w2,mw2,mb2,nw2,b2]
    // rgb  params base 89 + idx*4 : [w,mw,mb,b]
    // noises base 117 + idx*2

    // ---- workspace carve (256B aligned) ----
    char* wp = (char*)d_ws;
    auto carve = [&](size_t bytes) -> void* {
        void* ptr = (void*)wp;
        wp += (bytes + 255) & ~(size_t)255;
        return ptr;
    };
    float*     wlat = (float*)carve((size_t)BATCH * 512 * 4);
    float*     sbuf = (float*)carve((size_t)BATCH * 512 * 4);
    float*     dmod = (float*)carve((size_t)BATCH * 512 * 4);
    _Float16*  wh   = (_Float16*)carve((size_t)9 * 512 * 512 * 2);            // 4.7 MB
    _Float16*  xs   = (_Float16*)carve((size_t)BATCH * 258 * 258 * 128 * 2);  // 68 MB
    float*     xa   = (float*)carve((size_t)BATCH * 64 * 256 * 256 * 4);      // 67 MB
    float*     xb   = (float*)carve((size_t)BATCH * 64 * 256 * 256 * 4);      // 67 MB
    float*     rgbA = (float*)carve((size_t)BATCH * 3 * 256 * 256 * 4);
    float*     rgbB = (float*)carve((size_t)BATCH * 3 * 256 * 256 * 4);

    // ---- mapping network ----
    mapping_kernel<<<1, 512, 0, stream>>>(z, cf, ma, wlat);

    float* rgbPrev = nullptr;
    int in_ch = 512;
    for (int idx = 0; idx < 7; ++idx) {
        int r = RESV[idx], out_ch = CHV[idx];
        int npix = r * r;
        int cbase = 19 + idx * 10;
        int gbase = 89 + idx * 4;
        int nbase = 117 + idx * 2;

        // ---------- conv 1 ----------
        {
            int I = in_ch, O = out_ch;
            const float* w   = (const float*)d_in[cbase + 0];
            const float* mw  = (const float*)d_in[cbase + 1];
            const float* mb  = (const float*)d_in[cbase + 2];
            const float* nwp = (const float*)d_in[cbase + 3];
            const float* bs  = (const float*)d_in[cbase + 4];
            const float* nz  = (const float*)d_in[nbase + 0];

            style_kernel<<<(I + 255) / 256, 256, 0, stream>>>(wlat, mw, mb, sbuf, I);
            demod_kernel<<<(BATCH * O + 255) / 256, 256, 0, stream>>>(sbuf, w, dmod, I, O);
            prep_w_kernel<<<(O * I * 9 + 255) / 256, 256, 0, stream>>>(w, wh, I, O);
            long total = (long)BATCH * (r + 2) * (r + 2) * I;
            modulate_pad_kernel<<<(int)((total + 255) / 256), 256, 0, stream>>>(
                xa, constant, sbuf, xs, I, r, r, /*ups=*/(idx > 0), /*from_const=*/(idx == 0));
            launch_conv(xs, wh, dmod, nz, nwp, bs, xb, I, O, r, stream);
        }
        // ---------- conv 2 ----------
        {
            int I = out_ch, O = out_ch;
            const float* w   = (const float*)d_in[cbase + 5];
            const float* mw  = (const float*)d_in[cbase + 6];
            const float* mb  = (const float*)d_in[cbase + 7];
            const float* nwp = (const float*)d_in[cbase + 8];
            const float* bs  = (const float*)d_in[cbase + 9];
            const float* nz  = (const float*)d_in[nbase + 1];

            style_kernel<<<(I + 255) / 256, 256, 0, stream>>>(wlat, mw, mb, sbuf, I);
            demod_kernel<<<(BATCH * O + 255) / 256, 256, 0, stream>>>(sbuf, w, dmod, I, O);
            prep_w_kernel<<<(O * I * 9 + 255) / 256, 256, 0, stream>>>(w, wh, I, O);
            long total = (long)BATCH * (r + 2) * (r + 2) * I;
            modulate_pad_kernel<<<(int)((total + 255) / 256), 256, 0, stream>>>(
                xb, nullptr, sbuf, xs, I, r, r, /*ups=*/0, /*from_const=*/0);
            launch_conv(xs, wh, dmod, nz, nwp, bs, xa, I, O, r, stream);
        }
        // ---------- to_rgb ----------
        {
            int O = out_ch;
            const float* wr  = (const float*)d_in[gbase + 0];
            const float* mw  = (const float*)d_in[gbase + 1];
            const float* mb  = (const float*)d_in[gbase + 2];
            const float* br  = (const float*)d_in[gbase + 3];
            style_kernel<<<(O + 255) / 256, 256, 0, stream>>>(wlat, mw, mb, sbuf, O);
            float* rgbOut = (idx & 1) ? rgbB : rgbA;
            long total = (long)BATCH * npix;
            torgb_kernel<<<(int)((total + 255) / 256), 256, 0, stream>>>(
                xa, sbuf, wr, br, rgbPrev, rgbOut, O, r, r, /*has_skip=*/(idx > 0));
            rgbPrev = rgbOut;
        }
        in_ch = out_ch;
    }

    long n = (long)BATCH * 3 * 256 * 256;
    tanh_kernel<<<(int)((n + 255) / 256), 256, 0, stream>>>(rgbPrev, (float*)d_out, n);
}
